// moe_39616778338643
// MI455X (gfx1250) — matrix operand order repaired and checked by reading, not compiled, not run on hardware
//
#include <hip/hip_runtime.h>

#define NN     10000
#define DEG    16
#define EE     160000
#define DD     256
#define HH     128
#define NEXP   4
#define EDIM   8
#define INDIM  520
#define KC1    17          // ceil(544/32) K-chunks for layer 1 (IN_DIM padded 520->544)
#define KC2    4           // 128/32 K-chunks for layer 2
#define NT     8           // 128/16 N-tiles
#define TPB    10          // tile-groups per block in MLP kernel
#define NBLK_SORT 157      // ceil(160000/1024)

typedef __attribute__((ext_vector_type(16))) __bf16   v16bf;
typedef __attribute__((ext_vector_type(8)))  float    v8f;
typedef __attribute__((ext_vector_type(4)))  float    float4v;
typedef __attribute__((ext_vector_type(4)))  unsigned uint4v;
typedef __attribute__((ext_vector_type(4)))  unsigned u32x4;
typedef __attribute__((ext_vector_type(8)))  int      i32x8;
typedef __attribute__((ext_vector_type(4)))  int      i32x4;

__device__ __forceinline__ unsigned short f2bf(float f) {
  unsigned u = __float_as_uint(f);
  unsigned r = u + 0x7FFFu + ((u >> 16) & 1u);   // RNE
  return (unsigned short)(r >> 16);
}

// ---------------- init ----------------
__global__ void k_zero(float* gated, int* ncnt, float* sums) {
  int i = blockIdx.x * blockDim.x + threadIdx.x;
  if (i < EE) gated[i] = 0.f;
  if (i < NN) ncnt[i] = 0;
  if (i < 16) sums[i] = 0.f;
}

// ---------------- noisy top-k gating: one wave32 per node ----------------
__global__ void k_gate(const float* __restrict__ x, const float* __restrict__ wg,
                       const float* __restrict__ wn, const float* __restrict__ noise,
                       const float* __restrict__ klist,
                       float* __restrict__ gates, int* __restrict__ kmax,
                       float* __restrict__ sums) {
  int node = (blockIdx.x * blockDim.x + threadIdx.x) >> 5;
  int lane = threadIdx.x & 31;
  if (node >= NN) return;
  const float* xr = x + (size_t)node * DD;
  float cg0=0,cg1=0,cg2=0,cg3=0, cn0=0,cn1=0,cn2=0,cn3=0;
  for (int k = lane; k < DD; k += 32) {
    float xv = xr[k];
    float4v g = *(const float4v*)(wg + k*4);
    float4v n = *(const float4v*)(wn + k*4);
    cg0 += xv*g.x; cg1 += xv*g.y; cg2 += xv*g.z; cg3 += xv*g.w;
    cn0 += xv*n.x; cn1 += xv*n.y; cn2 += xv*n.z; cn3 += xv*n.w;
  }
  for (int off = 16; off; off >>= 1) {
    cg0 += __shfl_down(cg0, off, 32); cg1 += __shfl_down(cg1, off, 32);
    cg2 += __shfl_down(cg2, off, 32); cg3 += __shfl_down(cg3, off, 32);
    cn0 += __shfl_down(cn0, off, 32); cn1 += __shfl_down(cn1, off, 32);
    cn2 += __shfl_down(cn2, off, 32); cn3 += __shfl_down(cn3, off, 32);
  }
  if (lane == 0) {
    float clean[4] = {cg0, cg1, cg2, cg3};
    float cn[4]    = {cn0, cn1, cn2, cn3};
    float sd[4], nsy[4];
    #pragma unroll
    for (int e = 0; e < 4; ++e) {
      float sp = cn[e] > 20.f ? cn[e] : log1pf(expf(cn[e]));
      sd[e]  = sp + 0.01f;
      nsy[e] = clean[e] + noise[node*4 + e] * sd[e];
    }
    float l[4] = {nsy[0], nsy[1], nsy[2], nsy[3]};
    int  id[4] = {0, 1, 2, 3};
    #define CSW(a,b) if (l[a] < l[b]) { float tf=l[a];l[a]=l[b];l[b]=tf; int ti=id[a];id[a]=id[b];id[b]=ti; }
    CSW(0,1) CSW(2,3) CSW(0,2) CSW(1,3) CSW(1,2)
    #undef CSW
    float g0 = 1.f / (1.f + expf(l[1] - l[0]));
    float g1 = 1.f - g0;
    float gv[4] = {0.f, 0.f, 0.f, 0.f};
    gv[id[0]] = g0; gv[id[1]] = g1;
    float thrin = l[2], throut = l[1];
    #pragma unroll
    for (int e = 0; e < 4; ++e) {
      gates[node*4 + e] = gv[e];
      atomicAdd(&sums[e], gv[e]);
      float thr = (nsy[e] > thrin) ? thrin : throut;
      float z = (clean[e] - thr) / sd[e];
      atomicAdd(&sums[4 + e], 0.5f * erfcf(-z * 0.70710678118654752f));
    }
    float kpn = gv[0]*klist[0] + gv[1]*klist[1] + gv[2]*klist[2] + gv[3]*klist[3];
    int kl_ = (int)rintf(kpn * (float)DEG);
    if (kl_ < 1) kl_ = 1;
    if (kl_ > DEG) kl_ = DEG;
    kmax[node] = kl_;
  }
}

// ---------------- weight conversion: fp32 -> bf16, WMMA B-fragment layout ----------------
// B (32x16 bf16) packing here (contiguous k, as A): VGPR j half p: K = (lane<16?0:16)+2j+p, N = lane&15
__global__ void k_wconv1(const float* __restrict__ W1, unsigned short* __restrict__ w1f) {
  int idx = blockIdx.x * blockDim.x + threadIdx.x;
  if (idx >= NEXP*KC1*NT*512) return;
  int half = idx & 15, lane = (idx >> 4) & 31, frag = idx >> 9;
  int expert = frag / (KC1*NT);
  int rem = frag - expert*(KC1*NT);
  int c = rem >> 3, t = rem & 7;
  int K = c*32 + ((lane < 16) ? 0 : 16) + (half & ~1) + (half & 1);
  int Nc = t*16 + (lane & 15);
  float v = (K < INDIM) ? W1[((size_t)expert*INDIM + K)*HH + Nc] : 0.f;
  w1f[(size_t)frag*512 + lane*16 + half] = f2bf(v);
}
__global__ void k_wconv2(const float* __restrict__ W2, unsigned short* __restrict__ w2f) {
  int idx = blockIdx.x * blockDim.x + threadIdx.x;
  if (idx >= NEXP*KC2*NT*512) return;
  int half = idx & 15, lane = (idx >> 4) & 31, frag = idx >> 9;
  int expert = frag / (KC2*NT);
  int rem = frag - expert*(KC2*NT);
  int c = rem >> 3, t = rem & 7;
  int K = c*32 + ((lane < 16) ? 0 : 16) + (half & ~1) + (half & 1);
  int Nc = t*16 + (lane & 15);
  w2f[(size_t)frag*512 + lane*16 + half] = f2bf(W2[((size_t)expert*HH + K)*HH + Nc]);
}

// ---------------- expert MLP: WMMA bf16, weights resident in 320KB WGP LDS ----------------
__global__ void k_mlp(const float* __restrict__ x, const int* __restrict__ ei,
                      const float* __restrict__ eattr,
                      const float* __restrict__ b1, const float* __restrict__ b2,
                      const float* __restrict__ Wo, const float* __restrict__ bo,
                      const int* __restrict__ temp, const float* __restrict__ gates,
                      const unsigned short* __restrict__ w1f,
                      const unsigned short* __restrict__ w2f,
                      float* __restrict__ gated) {
  extern __shared__ unsigned char smem[];
  const int expert = blockIdx.y;
  unsigned short* w1s = (unsigned short*)smem;          // 17*8*512 ush = 139264 B
  unsigned short* w2s = w1s + KC1*NT*512;               // 4*8*512  ush =  32768 B
  float* b1s = (float*)(w2s + KC2*NT*512);              // 512 B
  float* b2s = b1s + HH;                                // 512 B
  float* wos = b2s + HH;                                // 512 B
  __bf16* hbase = (__bf16*)(wos + HH);                  // 8 waves * 16*128 bf16 = 32768 B

  // ---- stage W1 via Tensor Data Mover (one 139KB 1-D descriptor, wave 0 issues) ----
  if (threadIdx.x < 32) {
    unsigned lds_w1 = (unsigned)(size_t)(void*)w1s;     // generic addr low 32b == LDS offset
    unsigned long long ga =
        (unsigned long long)(size_t)(const void*)(w1f + (size_t)expert * (KC1*NT*512));
    // D# group0: count=1 | lds_addr | global_addr[56:0] | type=2
    u32x4 g0 = { 1u, lds_w1, (unsigned)ga, (unsigned)(ga >> 32) | (2u << 30) };
    // D# group1: data_size=4B; tensor_dim0 = tile_dim0 = tensor_dim0_stride = 34816 dwords
    i32x8 g1 = { (int)0x00020000,          // wg_mask=0, data_size=2(4B), no flags
                 (int)0x88000000u,         // tensor_dim0[15:0]=0x8800 in [31:16]
                 0,                        // tensor_dim0[31:16]=0, tensor_dim1[15:0]=0
                 (int)0x88000000u,         // tile_dim0=0x8800 in [31:16]
                 0,                        // tile_dim1=0, tile_dim2=0 (1-D tile)
                 (int)34816,               // tensor_dim0_stride[31:0]
                 0, 0 };
    i32x4 gz4 = { 0, 0, 0, 0 };
    i32x8 gz8 = { 0, 0, 0, 0, 0, 0, 0, 0 };
    __builtin_amdgcn_tensor_load_to_lds(g0, g1, gz4, gz4, gz8, 0);
    __builtin_amdgcn_s_wait_tensorcnt(0);
  }
  // ---- stage W2 via async global->LDS loads (ASYNCcnt path) ----
  {
    const unsigned short* s2 = w2f + (size_t)expert * (KC2*NT*512);
    for (int i = threadIdx.x * 8; i < KC2*NT*512; i += blockDim.x * 8) {
      unsigned lds = (unsigned)(size_t)(void*)(w2s + i);
      const void* gp = (const void*)(s2 + i);
      asm volatile("global_load_async_to_lds_b128 %0, %1, off"
                   :: "v"(lds), "v"(gp) : "memory");
    }
    asm volatile("s_wait_asynccnt 0x0" ::: "memory");
  }
  for (int i = threadIdx.x; i < HH; i += blockDim.x) {
    b1s[i] = b1[expert*HH + i];
    b2s[i] = b2[expert*HH + i];
    wos[i] = Wo[expert*HH + i];
  }
  __syncthreads();

  const float bo_e = bo[expert];
  const float tinv = 1.f / (float)temp[0];
  const int wave = threadIdx.x >> 5;
  const int lane = threadIdx.x & 31;
  const int row = lane & 15, grp = lane >> 4;
  __bf16* hb = hbase + wave * (16 * HH);

  union AU { v16bf v; unsigned short u[16]; uint4v q[2]; };

  for (int g = 0; g < TPB; ++g) {
    const int tile = (blockIdx.x * TPB + g) * 8 + wave;   // exactly covers 10000 tiles
    const int e = tile * 16 + row;
    const int s = ei[e], d = ei[EE + e];
    const float* xs = x + (size_t)s * DD;
    const float* xd = x + (size_t)d * DD;
    const float* ea = eattr + (size_t)e * EDIM;
    __builtin_prefetch(xs, 0, 0);
    __builtin_prefetch(xd, 0, 0);

    // ---- layer 1: [16 x 544] @ [544 x 128] via 17x8 WMMA bf16 ----
    v8f acc[NT];
    #pragma unroll
    for (int t = 0; t < NT; ++t) acc[t] = (v8f){0,0,0,0,0,0,0,0};
    for (int c = 0; c < KC1; ++c) {
      AU a;
      if (c < 16) {
        // A layout (16-bit 16x32): halves 0..15 -> K = grp*16+hh (matches B packing)
        const float* base = (c < 8) ? (xs + c*32) : (xd + (c - 8)*32);
        float4v q0 = *(const float4v*)(base + grp*16);
        float4v q1 = *(const float4v*)(base + grp*16 + 4);
        float4v q2 = *(const float4v*)(base + grp*16 + 8);
        float4v q3 = *(const float4v*)(base + grp*16 + 12);
        a.v[0]=(__bf16)q0.x; a.v[1]=(__bf16)q0.y; a.v[2]=(__bf16)q0.z; a.v[3]=(__bf16)q0.w;
        a.v[4]=(__bf16)q1.x; a.v[5]=(__bf16)q1.y; a.v[6]=(__bf16)q1.z; a.v[7]=(__bf16)q1.w;
        a.v[8]=(__bf16)q2.x; a.v[9]=(__bf16)q2.y; a.v[10]=(__bf16)q2.z; a.v[11]=(__bf16)q2.w;
        a.v[12]=(__bf16)q3.x; a.v[13]=(__bf16)q3.y; a.v[14]=(__bf16)q3.z; a.v[15]=(__bf16)q3.w;
      } else { // K 512..543: only K 512..519 (edge_attr) nonzero, grp 0, halves 0..7
        #pragma unroll
        for (int hh = 0; hh < 16; ++hh) a.u[hh] = 0;
        if (grp == 0) {
          float4v q0 = *(const float4v*)ea;
          float4v q1 = *(const float4v*)(ea + 4);
          a.v[0]=(__bf16)q0.x; a.v[1]=(__bf16)q0.y; a.v[2]=(__bf16)q0.z; a.v[3]=(__bf16)q0.w;
          a.v[4]=(__bf16)q1.x; a.v[5]=(__bf16)q1.y; a.v[6]=(__bf16)q1.z; a.v[7]=(__bf16)q1.w;
        }
      }
      // batch all 8 B fragments first so the 16 ds_load_b128 pipeline ahead of the WMMAs
      const unsigned char* wb = (const unsigned char*)w1s + ((c*NT) << 10) + (lane << 5);
      v16bf bfr[NT];
      #pragma unroll
      for (int t = 0; t < NT; ++t) bfr[t] = *(const v16bf*)(wb + (t << 10));
      #pragma unroll
      for (int t = 0; t < NT; ++t)
        acc[t] = __builtin_amdgcn_wmma_f32_16x16x32_bf16(false, a.v, false, bfr[t],
                                                         (short)0, acc[t], false, false);
    }
    // bias + relu -> h1 (bf16, row-major [16][128]) in private LDS tile
    #pragma unroll
    for (int t = 0; t < NT; ++t) {
      int ncol = t*16 + row;          // C layout: VGPR r, M = r + 8*grp, N = lane&15 (+16t)
      float bb = b1s[ncol];
      #pragma unroll
      for (int r = 0; r < 8; ++r) {
        float v = acc[t][r] + bb;
        hb[(r + grp*8)*HH + ncol] = (__bf16)(v > 0.f ? v : 0.f);
      }
    }
    asm volatile("s_wait_dscnt 0x0" ::: "memory");  // wave32 lockstep: LDS now visible

    // ---- layer 2: [16 x 128] @ [128 x 128] via 4x8 WMMA bf16 ----
    v8f acc2[NT];
    #pragma unroll
    for (int t = 0; t < NT; ++t) acc2[t] = (v8f){0,0,0,0,0,0,0,0};
    for (int c = 0; c < KC2; ++c) {
      AU a;
      const __bf16* hr = hb + row*HH + c*32 + grp*16;
      a.q[0] = *(const uint4v*)hr;          // halves 0..7
      a.q[1] = *(const uint4v*)(hr + 8);    // halves 8..15
      const unsigned char* wb = (const unsigned char*)w2s + ((c*NT) << 10) + (lane << 5);
      v16bf bfr[NT];
      #pragma unroll
      for (int t = 0; t < NT; ++t) bfr[t] = *(const v16bf*)(wb + (t << 10));
      #pragma unroll
      for (int t = 0; t < NT; ++t)
        acc2[t] = __builtin_amdgcn_wmma_f32_16x16x32_bf16(false, a.v, false, bfr[t],
                                                          (short)0, acc2[t], false, false);
    }
    #pragma unroll
    for (int t = 0; t < NT; ++t) {
      int ncol = t*16 + row;
      float bb = b2s[ncol];
      #pragma unroll
      for (int r = 0; r < 8; ++r) {
        float v = acc2[t][r] + bb;
        hb[(r + grp*8)*HH + ncol] = (__bf16)(v > 0.f ? v : 0.f);   // h2 overwrites h1
      }
    }
    asm volatile("s_wait_dscnt 0x0" ::: "memory");

    // ---- layer 3: per-edge dot(h2, Wo), split across the two half-wave groups ----
    float sum = 0.f;
    const __bf16* h2r = hb + row*HH + grp*64;
    const float* wor = wos + grp*64;
    #pragma unroll 8
    for (int n = 0; n < 64; ++n) sum += (float)h2r[n] * wor[n];
    sum += __shfl_xor(sum, 16, 32);
    if (grp == 0) {
      float z = (sum + bo_e) * tinv;
      float sig = 1.f / (1.f + expf(-z));
      float gw = gates[(size_t)s*NEXP + expert];
      atomicAdd(&gated[e], 0.25f * gw * sig);   // mean over NE experts
    }
  }
}

// ---------------- loss ----------------
__global__ void k_loss(const float* __restrict__ sums, float* __restrict__ out) {
  if (blockIdx.x != 0 || threadIdx.x != 0) return;
  float mi = 0.f, ml = 0.f;
  for (int e = 0; e < 4; ++e) { mi += sums[e]; ml += sums[4 + e]; }
  mi *= 0.25f; ml *= 0.25f;
  float vi = 0.f, vl = 0.f;
  for (int e = 0; e < 4; ++e) {
    float a = sums[e] - mi;     vi += a*a;
    float b = sums[4+e] - ml;   vl += b*b;
  }
  vi /= 3.f; vl /= 3.f;   // unbiased ddof=1
  out[EE] = 0.01f * (vi / (mi*mi + 1e-10f) + vl / (ml*ml + 1e-10f));
}

// ---------------- per-node threshold ----------------
__global__ void k_build(const int* __restrict__ ei, int* __restrict__ ncnt,
                        int* __restrict__ nedges) {
  int e = blockIdx.x * blockDim.x + threadIdx.x;
  if (e >= EE) return;
  int s = ei[e];
  int slot = atomicAdd(&ncnt[s], 1);
  nedges[s*DEG + slot] = e;
}
__global__ void k_thre(const float* __restrict__ gated, const int* __restrict__ nedges,
                       const int* __restrict__ kmax, float* __restrict__ nthre) {
  int i = blockIdx.x * blockDim.x + threadIdx.x;
  if (i >= NN) return;
  float sv[DEG];
  for (int j = 0; j < DEG; ++j) sv[j] = gated[nedges[i*DEG + j]];
  for (int a = 1; a < DEG; ++a) {          // insertion sort, descending
    float v = sv[a]; int b = a - 1;
    while (b >= 0 && sv[b] < v) { sv[b+1] = sv[b]; --b; }
    sv[b+1] = v;
  }
  int k = kmax[i]; if (k < 1) k = 1; if (k > DEG) k = DEG;
  nthre[i] = sv[k - 1];
}
__global__ void k_mask(const float* __restrict__ gated, const int* __restrict__ ei,
                       const float* __restrict__ nthre, float* __restrict__ m) {
  int e = blockIdx.x * blockDim.x + threadIdx.x;
  if (e >= EE) return;
  m[e] = (gated[e] - nthre[ei[e]] + 1e-12f > 0.f) ? 1.f : 0.f;
}

// ---------------- stable LSD radix argsort (descending by value) ----------------
__global__ void k_keys(const float* __restrict__ gated, unsigned* __restrict__ key,
                       int* __restrict__ idx) {
  int e = blockIdx.x * blockDim.x + threadIdx.x;
  if (e >= EE) return;
  unsigned u = __float_as_uint(gated[e]);
  unsigned ord = (u & 0x80000000u) ? ~u : (u | 0x80000000u);
  key[e] = ~ord;    // ascending sort of key == descending by value, stable ties by index
  idx[e] = e;
}
__global__ void k_hist(const unsigned* __restrict__ key, unsigned* __restrict__ bhist,
                       int shift) {
  __shared__ unsigned hc[256];
  hc[threadIdx.x] = 0;
  __syncthreads();
  int base = blockIdx.x * 1024 + threadIdx.x * 4;
  for (int i = 0; i < 4; ++i) {
    int e = base + i;
    if (e < EE) atomicAdd(&hc[(key[e] >> shift) & 255u], 1u);
  }
  __syncthreads();
  bhist[blockIdx.x * 256 + threadIdx.x] = hc[threadIdx.x];
}
__global__ void k_scan(unsigned* __restrict__ bhist) {
  __shared__ unsigned tot[256], st[256];
  int d = threadIdx.x;
  unsigned run = 0;
  for (int b = 0; b < NBLK_SORT; ++b) {
    unsigned v = bhist[b*256 + d];
    bhist[b*256 + d] = run;
    run += v;
  }
  tot[d] = run;
  __syncthreads();
  if (d == 0) {
    unsigned s = 0;
    for (int i = 0; i < 256; ++i) { st[i] = s; s += tot[i]; }
  }
  __syncthreads();
  unsigned s0 = st[d];
  for (int b = 0; b < NBLK_SORT; ++b) bhist[b*256 + d] += s0;
}
__global__ void k_scatter(const unsigned* __restrict__ keyIn, const int* __restrict__ idxIn,
                          unsigned* __restrict__ keyOut, int* __restrict__ idxOut,
                          const unsigned* __restrict__ bofs, int shift) {
  __shared__ unsigned base[256];
  base[threadIdx.x] = bofs[blockIdx.x * 256 + threadIdx.x];
  __syncthreads();
  int start = blockIdx.x * 1024;
  for (int t = 0; t < 256; ++t) {          // serialize for stability (not the bottleneck)
    if ((int)threadIdx.x == t) {
      for (int i = 0; i < 4; ++i) {
        int e = start + t*4 + i;
        if (e < EE) {
          unsigned k = keyIn[e];
          unsigned pos = base[(k >> shift) & 255u]++;
          keyOut[pos] = k;
          idxOut[pos] = idxIn[e];
        }
      }
    }
    __syncthreads();
  }
}

// out[j] = m[order[order[j]]]  (faithful to the reference's double index_select)
__global__ void k_final(const float* __restrict__ m, const int* __restrict__ ord,
                        float* __restrict__ out) {
  int j = blockIdx.x * blockDim.x + threadIdx.x;
  if (j >= EE) return;
  out[j] = m[ord[ord[j]]];
}

extern "C" void kernel_launch(void* const* d_in, const int* in_sizes, int n_in,
                              void* d_out, int out_size, void* d_ws, size_t ws_size,
                              hipStream_t stream) {
  (void)in_sizes; (void)n_in; (void)out_size; (void)ws_size;
  const float* x   = (const float*)d_in[0];
  const int*   ei  = (const int*)d_in[1];
  const float* ea  = (const float*)d_in[2];
  const float* noi = (const float*)d_in[3];
  const float* wg  = (const float*)d_in[4];
  const float* wn  = (const float*)d_in[5];
  const float* W1  = (const float*)d_in[6];
  const float* b1  = (const float*)d_in[7];
  const float* W2  = (const float*)d_in[8];
  const float* b2  = (const float*)d_in[9];
  const float* Wo  = (const float*)d_in[10];
  const float* bo  = (const float*)d_in[11];
  const float* kl  = (const float*)d_in[12];
  const int*   tmp = (const int*)d_in[13];
  float* out = (float*)d_out;

  unsigned char* ws = (unsigned char*)d_ws;
  size_t off = 0;
  auto alloc = [&](size_t bytes) -> void* {
    off = (off + 255) & ~(size_t)255;
    void* p = ws + off;
    off += bytes;
    return p;
  };
  float* gates  = (float*)alloc((size_t)NN * NEXP * 4);
  int*   kmax   = (int*)  alloc((size_t)NN * 4);
  float* gated  = (float*)alloc((size_t)EE * 4);
  float* mvals  = (float*)alloc((size_t)EE * 4);
  float* nthre  = (float*)alloc((size_t)NN * 4);
  int*   ncnt   = (int*)  alloc((size_t)NN * 4);
  int*   nedges = (int*)  alloc((size_t)NN * DEG * 4);
  unsigned* keyA = (unsigned*)alloc((size_t)EE * 4);
  unsigned* keyB = (unsigned*)alloc((size_t)EE * 4);
  int*   idxA   = (int*)  alloc((size_t)EE * 4);
  int*   idxB   = (int*)  alloc((size_t)EE * 4);
  unsigned* bhist = (unsigned*)alloc((size_t)NBLK_SORT * 256 * 4);
  float* sums   = (float*)alloc(16 * 4);
  unsigned short* w1f = (unsigned short*)alloc((size_t)NEXP * KC1 * NT * 512 * 2);
  unsigned short* w2f = (unsigned short*)alloc((size_t)NEXP * KC2 * NT * 512 * 2);

  k_zero<<<625, 256, 0, stream>>>(gated, ncnt, sums);
  k_gate<<<1250, 256, 0, stream>>>(x, wg, wn, noi, kl, gates, kmax, sums);
  k_wconv1<<<(NEXP*KC1*NT*512 + 255)/256, 256, 0, stream>>>(W1, w1f);
  k_wconv2<<<(NEXP*KC2*NT*512 + 255)/256, 256, 0, stream>>>(W2, w2f);

  dim3 mlpGrid(125, NEXP);
  size_t smemBytes = (size_t)KC1*NT*1024 + (size_t)KC2*NT*1024 + 3*HH*4 + 8*16*HH*2; // 206336
  k_mlp<<<mlpGrid, 256, smemBytes, stream>>>(x, ei, ea, b1, b2, Wo, bo, tmp,
                                             gates, w1f, w2f, gated);

  k_loss<<<1, 32, 0, stream>>>(sums, out);
  k_build<<<625, 256, 0, stream>>>(ei, ncnt, nedges);
  k_thre<<<(NN + 255)/256, 256, 0, stream>>>(gated, nedges, kmax, nthre);
  k_mask<<<625, 256, 0, stream>>>(gated, ei, nthre, mvals);
  k_keys<<<625, 256, 0, stream>>>(gated, keyA, idxA);
  for (int p = 0; p < 4; ++p) {
    int shift = p * 8;
    const unsigned* ki = (p & 1) ? keyB : keyA;
    const int*      ii = (p & 1) ? idxB : idxA;
    unsigned*       ko = (p & 1) ? keyA : keyB;
    int*            io = (p & 1) ? idxA : idxB;
    k_hist<<<NBLK_SORT, 256, 0, stream>>>(ki, bhist, shift);
    k_scan<<<1, 256, 0, stream>>>(bhist);
    k_scatter<<<NBLK_SORT, 256, 0, stream>>>(ki, ii, ko, io, bhist, shift);
  }
  k_final<<<625, 256, 0, stream>>>(mvals, idxA, out);
}
